// DecoderOnlyTransformer_52441550684712
// MI455X (gfx1250) — matrix operand order repaired and checked by reading, not compiled, not run on hardware
//
#include <hip/hip_runtime.h>
#include <hip/hip_bf16.h>
#include <math.h>

// ---------- CDNA5 WMMA types ----------
typedef __attribute__((ext_vector_type(16))) __bf16 v16bf;
typedef __attribute__((ext_vector_type(8)))  float  v8f;
typedef __attribute__((ext_vector_type(4)))  unsigned int v4u;
typedef __attribute__((ext_vector_type(8)))  unsigned int v8u;

union Frag16 { v16bf v; unsigned int u[8]; };

static __device__ __forceinline__ unsigned short f2bfbits(float f) {
    unsigned int u = __float_as_uint(f);
    unsigned int r = u + 0x7FFFu + ((u >> 16) & 1u);   // round-to-nearest-even
    return (unsigned short)(r >> 16);
}
static __device__ __forceinline__ unsigned int pack2bf(float lo, float hi) {
    return (unsigned int)f2bfbits(lo) | ((unsigned int)f2bfbits(hi) << 16);
}

#define WMMA_BF16(A, B, C) \
    __builtin_amdgcn_wmma_f32_16x16x32_bf16(false, (A), false, (B), (short)0, (C), false, false)

// ==========================================================================
// Tensor Data Mover: DMA a [128 rows x 32 bf16] tile from global (row stride
// K elements) into LDS with 1-DWORD padding after every 16 DWORDs, i.e. an
// LDS row stride of 34 bf16 (17 u32) — matches the bank-conflict-free layout
// the WMMA fragment readers use. D# built per CDNA5 ISA.
// ==========================================================================
static __device__ __forceinline__ void tdm_load_a_tile(
    unsigned int lds_addr, unsigned long long gaddr, unsigned int K, unsigned int M)
{
    v4u g0; v8u g1;
    v4u g2 = {0u, 0u, 0u, 0u};
    v4u g3 = {0u, 0u, 0u, 0u};
    // group0: count=1 | lds_addr | global_addr[56:0] | type=2
    g0[0] = 1u;
    g0[1] = lds_addr;
    g0[2] = (unsigned int)gaddr;
    g0[3] = ((unsigned int)(gaddr >> 32) & 0x01FFFFFFu) | (2u << 30);
    // group1: data_size=1(2B) pad_enable pad_interval=3(16 DW) pad_amount=0(1 DW)
    g1[0] = (1u << 16) | (1u << 20) | (3u << 22);
    g1[1] = (K & 0xFFFFu) << 16;                        // tensor_dim0[15:0] @ bit48
    g1[2] = (K >> 16) | ((M & 0xFFFFu) << 16);          // tensor_dim0[31:16] | tensor_dim1[15:0]
    g1[3] = (M >> 16) | (32u << 16);                    // tensor_dim1[31:16] | tile_dim0=32
    g1[4] = 128u;                                       // tile_dim1=128, tile_dim2=0
    g1[5] = K;                                          // tensor_dim0_stride[31:0]
    g1[6] = 0u;                                         // stride[47:32]=0, dim1_stride lo=0
    g1[7] = 0u;
    asm volatile("tensor_load_to_lds %0, %1, %2, %3"
                 :: "s"(g0), "s"(g1), "s"(g2), "s"(g3)
                 : "memory");
}

// ==========================================================================
// Embedding + sinusoidal positional encoding -> fp32 residual stream
// ==========================================================================
__global__ __launch_bounds__(256) void embed_pos_kernel(
    const int* __restrict__ idx, const float* __restrict__ embed,
    float* __restrict__ x, int T, int D)
{
    int i = blockIdx.x * 256 + threadIdx.x;     // over T*D
    int t = i / D, d = i % D;
    int tok = idx[t];
    float freq = expf(-(float)(d & ~1) * (logf(10000.0f) / (float)D));
    float ang  = (float)t * freq;
    float pe   = (d & 1) ? cosf(ang) : sinf(ang);
    x[i] = embed[(size_t)tok * D + d] + pe;
}

// ==========================================================================
// LayerNorm: fp32 in -> bf16 out (GEMM A operand). One row (D=1024) / block.
// ==========================================================================
__global__ __launch_bounds__(256) void layernorm_bf16_kernel(
    const float* __restrict__ x, const float* __restrict__ g,
    const float* __restrict__ b, unsigned short* __restrict__ out, int D)
{
    __shared__ float red[256];
    const int row = blockIdx.x, tid = threadIdx.x;
    const float4 xv = *(const float4*)(x + (size_t)row * D + tid * 4);

    float s = xv.x + xv.y + xv.z + xv.w;
    red[tid] = s; __syncthreads();
    #pragma unroll
    for (int o = 128; o; o >>= 1) { if (tid < o) red[tid] += red[tid + o]; __syncthreads(); }
    float mu = red[0] / (float)D;
    __syncthreads();

    float dx0 = xv.x - mu, dx1 = xv.y - mu, dx2 = xv.z - mu, dx3 = xv.w - mu;
    red[tid] = dx0*dx0 + dx1*dx1 + dx2*dx2 + dx3*dx3; __syncthreads();
    #pragma unroll
    for (int o = 128; o; o >>= 1) { if (tid < o) red[tid] += red[tid + o]; __syncthreads(); }
    float rs = rsqrtf(red[0] / (float)D + 1e-5f);

    const float4 gv = *(const float4*)(g + tid * 4);
    const float4 bv = *(const float4*)(b + tid * 4);
    float y0 = dx0 * rs * gv.x + bv.x;
    float y1 = dx1 * rs * gv.y + bv.y;
    float y2 = dx2 * rs * gv.z + bv.z;
    float y3 = dx3 * rs * gv.w + bv.w;
    unsigned int* o32 = (unsigned int*)(out + (size_t)row * D + tid * 4);
    o32[0] = pack2bf(y0, y1);
    o32[1] = pack2bf(y2, y3);
}

// ==========================================================================
// GEMM: C[M,N] = A_bf16[M,K] @ B_f32->bf16[K,N] (+bias) (+residual) (gelu)
// Block: 256 threads (8 waves), tile 128x64, BK=32.
// Waves in 4(row)x2(col) grid; each wave owns a 32x32 output => 4 WMMAs per
// k-step from 2 A-frags x 2 B-frags (8 u32 LDS loads per WMMA).
// A tile is DMAed by the Tensor Data Mover into double-buffered LDS
// (software-pipelined: next tile's TDM overlaps current tile's math).
// B tile is fp32->bf16 converted in-register during staging.
// ==========================================================================
template<bool OUT_BF16, bool GELU, bool RESID>
__global__ __launch_bounds__(256) void gemm_wmma_kernel(
    const unsigned short* __restrict__ A, const float* __restrict__ B,
    const float* __restrict__ bias, const float* __restrict__ resid,
    void* __restrict__ out, int M, int N, int K)
{
    __shared__ unsigned int As[2][128 * 17];   // 128 rows x 34 bf16 (TDM-padded)
    __shared__ unsigned int Bs[32 * 33];       // 32 rows x 66 bf16

    const int tid  = threadIdx.x;
    const int lane = tid & 31, wave = tid >> 5;
    const int m0 = blockIdx.y * 128, n0 = blockIdx.x * 64;
    const int rb  = (wave & 3) * 32;           // wave's row block
    const int cbl = (wave >> 2) * 32;          // wave's col block (bf16 elems)
    const int half = lane >> 4, r16 = lane & 15;

    const int brow = tid >> 3, boc = tid & 7;  // B staging: 32 rows x 8 threads

    const unsigned long long aBase =
        (unsigned long long)(uintptr_t)A + ((size_t)m0 * K) * 2ull;
    const unsigned int asOff0 = (unsigned int)(uintptr_t)&As[0][0];
    const unsigned int asOff1 = (unsigned int)(uintptr_t)&As[1][0];

    v8f acc[2][2];
    #pragma unroll
    for (int ri = 0; ri < 2; ++ri)
        #pragma unroll
        for (int ci = 0; ci < 2; ++ci)
            #pragma unroll
            for (int r = 0; r < 8; ++r) acc[ri][ci][r] = 0.0f;

    // prologue: DMA first A tile
    if (wave == 0)
        tdm_load_a_tile(asOff0, aBase, (unsigned)K, (unsigned)M);

    int buf = 0;
    for (int kk = 0; kk < K; kk += 32, buf ^= 1) {
        // ---- stage B (fp32 -> bf16 convert) ----
        const float* bg = B + (size_t)(kk + brow) * N + n0 + boc * 8;
        const float4 b0 = *(const float4*)bg;
        const float4 b1 = *(const float4*)(bg + 4);
        unsigned int* bd = &Bs[brow * 33 + boc * 4];
        bd[0] = pack2bf(b0.x, b0.y); bd[1] = pack2bf(b0.z, b0.w);
        bd[2] = pack2bf(b1.x, b1.y); bd[3] = pack2bf(b1.z, b1.w);

        if (kk + 32 < K)   // prefetch next weight tile (global_prefetch path)
            __builtin_prefetch(B + (size_t)(kk + 32 + brow) * N + n0 + boc * 8, 0, 3);

        if (wave == 0) __builtin_amdgcn_s_wait_tensorcnt(0);
        __syncthreads();

        // pipeline: kick off next A tile's DMA into the other buffer
        if (wave == 0 && kk + 32 < K)
            tdm_load_a_tile(buf ? asOff0 : asOff1,
                            aBase + (unsigned long long)(kk + 32) * 2ull,
                            (unsigned)K, (unsigned)M);

        // ---- fragments ----
        Frag16 af[2];
        #pragma unroll
        for (int ri = 0; ri < 2; ++ri) {
            const unsigned int* arp = &As[buf][(rb + ri * 16 + r16) * 17];
            #pragma unroll
            for (int g = 0; g < 2; ++g)
                #pragma unroll
                for (int p = 0; p < 4; ++p)
                    af[ri].u[g * 4 + p] = arp[g * 8 + half * 4 + p];
        }
        Frag16 bfr[2];
        #pragma unroll
        for (int ci = 0; ci < 2; ++ci)
            #pragma unroll
            for (int j = 0; j < 8; ++j)
                bfr[ci].u[j] = (unsigned int)((const unsigned short*)Bs)[((j >> 2) * 16 + half * 8 + (j & 3) * 2) * 66 + cbl + ci * 16 + r16] | ((unsigned int)((const unsigned short*)Bs)[((j >> 2) * 16 + half * 8 + (j & 3) * 2 + 1) * 66 + cbl + ci * 16 + r16] << 16);

        #pragma unroll
        for (int ri = 0; ri < 2; ++ri)
            #pragma unroll
            for (int ci = 0; ci < 2; ++ci)
                acc[ri][ci] = WMMA_BF16(af[ri].v, bfr[ci].v, acc[ri][ci]);

        __syncthreads();
    }

    // ---- epilogue ----
    #pragma unroll
    for (int ri = 0; ri < 2; ++ri)
        #pragma unroll
        for (int ci = 0; ci < 2; ++ci) {
            const int ng = n0 + cbl + ci * 16 + r16;
            const float bvv = bias ? bias[ng] : 0.0f;
            #pragma unroll
            for (int r = 0; r < 8; ++r) {
                const int row = m0 + rb + ri * 16 + r + 8 * half;
                float v = acc[ri][ci][r] + bvv;
                if (RESID) v += resid[(size_t)row * N + ng];
                if (GELU)  v = 0.5f * v * (1.0f + erff(v * 0.70710678118f));
                if (OUT_BF16) ((unsigned short*)out)[(size_t)row * N + ng] = f2bfbits(v);
                else          ((float*)out)[(size_t)row * N + ng] = v;
            }
        }
}

// ==========================================================================
// Flash attention (causal, one head per block.y, 64 query rows per block.x)
// Block: 128 threads (4 waves); wave w owns query rows [qb0+16w, qb0+16w+16)
// Key blocks of 32; S = Q K^T via WMMA (2 n-tiles x 2 k-steps), online
// softmax with half-wave shuffle reductions, P@V via WMMA (4 d-tiles).
// ==========================================================================
__global__ __launch_bounds__(128) void flash_attn_kernel(
    const unsigned short* __restrict__ q, const unsigned short* __restrict__ k,
    const unsigned short* __restrict__ v, unsigned short* __restrict__ o,
    int D)
{
    __shared__ unsigned int qs [64 * 33];      // Q  [row][66 bf16]
    __shared__ unsigned int kts[64 * 17];      // K^T [d][34 bf16]
    __shared__ unsigned int vs [32 * 33];      // V  [key][66 bf16]
    __shared__ unsigned int ps [4 * 16 * 17];  // per-wave P 16x34 bf16

    const int tid = threadIdx.x, lane = tid & 31, wave = tid >> 5;
    const int h = blockIdx.y, qb0 = blockIdx.x * 64;
    const int qbase = qb0 + wave * 16;
    const int half = lane >> 4, r16 = lane & 15;
    const int colbase = h * 64;

    // ---- stage Q once ----
    for (int i = tid; i < 64 * 32; i += 128) {
        int row = i >> 5, cu = i & 31;
        qs[row * 33 + cu] =
            *(const unsigned int*)(q + (size_t)(qb0 + row) * D + colbase + cu * 2);
    }
    __syncthreads();

    Frag16 qf[2];
    {
        const unsigned int* qp = &qs[(wave * 16 + r16) * 33];
        #pragma unroll
        for (int g2 = 0; g2 < 2; ++g2)
            #pragma unroll
            for (int gg = 0; gg < 2; ++gg)
                #pragma unroll
                for (int p = 0; p < 4; ++p)
                    qf[g2].u[gg * 4 + p] = qp[g2 * 16 + gg * 8 + half * 4 + p];
    }

    v8f ofr[4];
    float mrow[8], lsum[8];
    #pragma unroll
    for (int t = 0; t < 4; ++t)
        #pragma unroll
        for (int r = 0; r < 8; ++r) ofr[t][r] = 0.0f;
    #pragma unroll
    for (int r = 0; r < 8; ++r) { mrow[r] = -3.0e38f; lsum[r] = 0.0f; }

    const int kmax = qb0 + 64;
    for (int kb = 0; kb < kmax; kb += 32) {
        __syncthreads();
        // ---- stage K (transposed) and V ----
        for (int i = tid; i < 32 * 32; i += 128) {
            int key = i >> 5, du = i & 31;
            unsigned int kw =
                *(const unsigned int*)(k + (size_t)(kb + key) * D + colbase + du * 2);
            unsigned short* k16 = (unsigned short*)kts;
            k16[(du * 2)     * 34 + key] = (unsigned short)(kw & 0xFFFFu);
            k16[(du * 2 + 1) * 34 + key] = (unsigned short)(kw >> 16);
            vs[key * 33 + du] =
                *(const unsigned int*)(v + (size_t)(kb + key) * D + colbase + du * 2);
        }
        __syncthreads();

        if (kb <= qbase + 15) {       // wave-uniform causal skip
            // ---- S = scale * Q K^T ----
            v8f s[2];
            #pragma unroll
            for (int nc = 0; nc < 2; ++nc) {
                #pragma unroll
                for (int r = 0; r < 8; ++r) s[nc][r] = 0.0f;
                Frag16 kf0, kf1;
                #pragma unroll
                for (int j = 0; j < 8; ++j) {
                    kf0.u[j] = (unsigned int)((const unsigned short*)kts)[((j >> 2) * 16 + half * 8 + (j & 3) * 2) * 34 + nc * 16 + r16] | ((unsigned int)((const unsigned short*)kts)[((j >> 2) * 16 + half * 8 + (j & 3) * 2 + 1) * 34 + nc * 16 + r16] << 16);
                    kf1.u[j] = (unsigned int)((const unsigned short*)kts)[(32 + (j >> 2) * 16 + half * 8 + (j & 3) * 2) * 34 + nc * 16 + r16] | ((unsigned int)((const unsigned short*)kts)[(32 + (j >> 2) * 16 + half * 8 + (j & 3) * 2 + 1) * 34 + nc * 16 + r16] << 16);
                }
                s[nc] = WMMA_BF16(qf[0].v, kf0.v, s[nc]);
                s[nc] = WMMA_BF16(qf[1].v, kf1.v, s[nc]);
            }
            // ---- scale + causal mask ----
            #pragma unroll
            for (int nc = 0; nc < 2; ++nc)
                #pragma unroll
                for (int r = 0; r < 8; ++r) {
                    float sv = s[nc][r] * 0.125f;
                    int ngl = kb + nc * 16 + r16;
                    int mgl = qbase + r + 8 * half;
                    s[nc][r] = (ngl > mgl) ? -3.0e38f : sv;
                }
            // ---- online softmax (half-wave reductions) ----
            float alpha[8];
            #pragma unroll
            for (int r = 0; r < 8; ++r) {
                float rm = fmaxf(s[0][r], s[1][r]);
                #pragma unroll
                for (int off = 8; off; off >>= 1)
                    rm = fmaxf(rm, __shfl_xor(rm, off, 32));
                float mn = fmaxf(mrow[r], rm);
                alpha[r] = expf(mrow[r] - mn);
                mrow[r]  = mn;
                float p0 = expf(s[0][r] - mn);
                float p1 = expf(s[1][r] - mn);
                s[0][r] = p0; s[1][r] = p1;
                float rs = p0 + p1;
                #pragma unroll
                for (int off = 8; off; off >>= 1)
                    rs += __shfl_xor(rs, off, 32);
                lsum[r] = lsum[r] * alpha[r] + rs;
            }
            #pragma unroll
            for (int t = 0; t < 4; ++t)
                #pragma unroll
                for (int r = 0; r < 8; ++r) ofr[t][r] *= alpha[r];

            // ---- relayout P (C-layout -> A-fragment) via wave-private LDS ----
            unsigned short* p16 = ((unsigned short*)ps) + wave * 16 * 34;
            #pragma unroll
            for (int r = 0; r < 8; ++r) {
                int m = r + 8 * half;
                p16[m * 34 + r16]      = f2bfbits(s[0][r]);
                p16[m * 34 + 16 + r16] = f2bfbits(s[1][r]);
            }
            Frag16 pf;
            const unsigned int* pp = &ps[wave * 16 * 17 + r16 * 17];
            #pragma unroll
            for (int gg = 0; gg < 2; ++gg)
                #pragma unroll
                for (int p = 0; p < 4; ++p)
                    pf.u[gg * 4 + p] = pp[gg * 8 + half * 4 + p];

            // ---- O += P @ V ----
            #pragma unroll
            for (int dt = 0; dt < 4; ++dt) {
                Frag16 vf;
                #pragma unroll
                for (int j = 0; j < 8; ++j) vf.u[j] = (unsigned int)((const unsigned short*)vs)[((j >> 2) * 16 + half * 8 + (j & 3) * 2) * 66 + dt * 16 + r16] | ((unsigned int)((const unsigned short*)vs)[((j >> 2) * 16 + half * 8 + (j & 3) * 2 + 1) * 66 + dt * 16 + r16] << 16);
                ofr[dt] = WMMA_BF16(pf.v, vf.v, ofr[dt]);
            }
        }
    }

    // ---- normalize + store ----
    #pragma unroll
    for (int r = 0; r < 8; ++r) {
        float inv = 1.0f / lsum[r];
        int row = qbase + r + 8 * half;
        #pragma unroll
        for (int dt = 0; dt < 4; ++dt)
            o[(size_t)row * D + colbase + dt * 16 + r16] =
                f2bfbits(ofr[dt][r] * inv);
    }
}

// ==========================================================================
// Host-side orchestration
// ==========================================================================
extern "C" void kernel_launch(void* const* d_in, const int* in_sizes, int n_in,
                              void* d_out, int out_size, void* d_ws, size_t ws_size,
                              hipStream_t stream)
{
    constexpr int T = 2048, D = 1024, H = 16, L = 4, Vv = 32000, D4 = 4096;

    const int*   idx    = (const int*)  d_in[0];
    const float* embed  = (const float*)d_in[1];
    const float* Wq     = (const float*)d_in[2];
    const float* Wk     = (const float*)d_in[3];
    const float* Wv     = (const float*)d_in[4];
    const float* Wo     = (const float*)d_in[5];
    const float* bq     = (const float*)d_in[6];
    const float* bk     = (const float*)d_in[7];
    const float* bv     = (const float*)d_in[8];
    const float* bo     = (const float*)d_in[9];
    const float* ln1_g  = (const float*)d_in[10];
    const float* ln1_b  = (const float*)d_in[11];
    const float* ln2_g  = (const float*)d_in[12];
    const float* ln2_b  = (const float*)d_in[13];
    const float* w1     = (const float*)d_in[14];
    const float* b1     = (const float*)d_in[15];
    const float* w2     = (const float*)d_in[16];
    const float* b2     = (const float*)d_in[17];
    const float* lnf_g  = (const float*)d_in[18];
    const float* lnf_b  = (const float*)d_in[19];
    const float* head_w = (const float*)d_in[20];

    char* ws = (char*)d_ws;
    size_t off = 0;
    float*          x0  = (float*)(ws + off);          off += (size_t)T * D * 4;
    float*          x1  = (float*)(ws + off);          off += (size_t)T * D * 4;
    unsigned short* hb  = (unsigned short*)(ws + off); off += (size_t)T * D * 2;
    unsigned short* qb_ = (unsigned short*)(ws + off); off += (size_t)T * D * 2;
    unsigned short* kb_ = (unsigned short*)(ws + off); off += (size_t)T * D * 2;
    unsigned short* vb_ = (unsigned short*)(ws + off); off += (size_t)T * D * 2;
    unsigned short* ob_ = (unsigned short*)(ws + off); off += (size_t)T * D * 2;
    unsigned short* ff  = (unsigned short*)(ws + off); off += (size_t)T * D4 * 2;
    (void)ws_size; (void)n_in; (void)in_sizes; (void)out_size;

    // x0 = embed[idx] + pos_encoding
    embed_pos_kernel<<<(T * D) / 256, 256, 0, stream>>>(idx, embed, x0, T, D);

    const dim3 gProj(D  / 64, T / 128);
    const dim3 gFF1 (D4 / 64, T / 128);
    const dim3 gHead(Vv / 64, T / 128);
    const dim3 gAttn(T / 64, H);

    for (int l = 0; l < L; ++l) {
        const size_t wOff = (size_t)l * D * D;
        // h = LN1(x0)
        layernorm_bf16_kernel<<<T, 256, 0, stream>>>(x0, ln1_g + l * D, ln1_b + l * D, hb, D);
        // q,k,v projections (bf16 out)
        gemm_wmma_kernel<true, false, false><<<gProj, 256, 0, stream>>>(
            hb, Wq + wOff, bq + l * D, nullptr, qb_, T, D, D);
        gemm_wmma_kernel<true, false, false><<<gProj, 256, 0, stream>>>(
            hb, Wk + wOff, bk + l * D, nullptr, kb_, T, D, D);
        gemm_wmma_kernel<true, false, false><<<gProj, 256, 0, stream>>>(
            hb, Wv + wOff, bv + l * D, nullptr, vb_, T, D, D);
        // causal attention -> o (bf16)
        flash_attn_kernel<<<gAttn, 128, 0, stream>>>(qb_, kb_, vb_, ob_, D);
        // x1 = x0 + o @ Wo + bo
        gemm_wmma_kernel<false, false, true><<<gProj, 256, 0, stream>>>(
            ob_, Wo + wOff, bo + l * D, x0, x1, T, D, D);
        // h = LN2(x1)
        layernorm_bf16_kernel<<<T, 256, 0, stream>>>(x1, ln2_g + l * D, ln2_b + l * D, hb, D);
        // ff = gelu(h @ w1 + b1)  (bf16 out)
        gemm_wmma_kernel<true, true, false><<<gFF1, 256, 0, stream>>>(
            hb, w1 + (size_t)l * D * D4, b1 + (size_t)l * D4, nullptr, ff, T, D4, D);
        // x0 = x1 + ff @ w2 + b2
        gemm_wmma_kernel<false, false, true><<<gProj, 256, 0, stream>>>(
            ff, w2 + (size_t)l * D4 * D, b2 + l * D, x1, x0, T, D, D4);
    }

    // final LN + LM head
    layernorm_bf16_kernel<<<T, 256, 0, stream>>>(x0, lnf_g, lnf_b, hb, D);
    gemm_wmma_kernel<false, false, false><<<gHead, 256, 0, stream>>>(
        hb, head_w, nullptr, nullptr, (float*)d_out, T, Vv, D);
}
